// MultiHeadAttention_15625091023113
// MI455X (gfx1250) — matrix operand order repaired and checked by reading, not compiled, not run on hardware
//
#include <hip/hip_runtime.h>
#include <hip/hip_bf16.h>

typedef _Float16 half_t;
typedef __attribute__((ext_vector_type(16))) _Float16 v16h;
typedef __attribute__((ext_vector_type(8)))  _Float16 v8h;
typedef __attribute__((ext_vector_type(8)))  float    v8f;

union V16 { v16h v; v8h h[2]; };

// ---- epilogue modes ----
#define EPI_F16_PLAIN  0   // f16 out[m*ldo+n]
#define EPI_F16_VHEAD  1   // f16 Pv written per-head transposed [64,2048]
#define EPI_F32_SCORES 2   // f32 out[z*strideCz + m*ldo + n] * scale
#define EPI_F16_CTX    3   // f16 ctx[b, s2, h*64+d]
#define EPI_F32_RES    4   // f32 out = D + residual

// ---------- async global->LDS 16B copy (gfx1250 ASYNCcnt path) ----------
#if defined(__has_builtin)
#if __has_builtin(__builtin_amdgcn_global_load_async_to_lds_b128)
#define ASYNC_BUILTIN 1
#endif
#if __has_builtin(__builtin_amdgcn_s_wait_asynccnt)
#define WAITASYNC_BUILTIN 1
#endif
#endif
#ifndef ASYNC_BUILTIN
#define ASYNC_BUILTIN 0
#endif
#ifndef WAITASYNC_BUILTIN
#define WAITASYNC_BUILTIN 0
#endif

typedef int vi4 __attribute__((vector_size(16)));
typedef __attribute__((address_space(1))) vi4* gas_vi4p;   // global int4*
typedef __attribute__((address_space(3))) vi4* las_vi4p;   // LDS int4*

__device__ inline void async_cp16B(void* lds, const void* g) {
#if ASYNC_BUILTIN
  __builtin_amdgcn_global_load_async_to_lds_b128(
      (gas_vi4p)(size_t)g,        // global addr == generic addr (value-preserving)
      (las_vi4p)lds,              // addrspacecast generic -> LDS
      0, 0);
#else
  unsigned lo = (unsigned)(size_t)(__attribute__((address_space(3))) void*)lds;
  unsigned long long ga = (unsigned long long)(size_t)g;
  asm volatile("global_load_async_to_lds_b128 %0, %1, off"
               :: "v"(lo), "v"(ga) : "memory");
#endif
}

__device__ inline void wait_asynccnt0() {
#if WAITASYNC_BUILTIN
  __builtin_amdgcn_s_wait_asynccnt(0);
#else
  asm volatile("s_wait_asynccnt 0x0" ::: "memory");
#endif
}

// ---------- register staging for fp32 sources (convert to f16) ----------
union StageChunk {
  float  f[8];
  float4 q[2];
  v8h    h;
};

__device__ inline void ld_stage(StageChunk& s, const float* p) {
  s.q[0] = *(const float4*)p;
  s.q[1] = *(const float4*)(p + 4);
}
__device__ inline void zero_stage(StageChunk& s) {
  const float4 z = make_float4(0.f, 0.f, 0.f, 0.f);
  s.q[0] = z; s.q[1] = z;
}
__device__ inline v8h to_h8(const StageChunk& s) {
  v8h r;
#pragma unroll
  for (int i = 0; i < 8; ++i) r[i] = (_Float16)s.f[i];
  return r;
}

// C[m,n] = sum_k A[m,k] * B[n,k]   (A row-major [M,K], B row-major [N,K])
// Block tile 128x128, BK=32, 256 threads = 8 waves (2x4), wave tile 64x32 (4x2 frags).
// Double-buffered LDS; f16 operands stream global->LDS via async copies,
// fp32 operands are VGPR-staged and converted to f16.
template <typename TA, typename TB, int EPI>
__global__ __launch_bounds__(256) void gemm_wmma(
    const TA* __restrict__ A, long lda, long strideAz,
    const TB* __restrict__ B, long ldb, long strideBz,
    void* __restrict__ Out, long ldo, long strideCz,
    const float* __restrict__ Res,
    int M, int N, int K, float scale)
{
  constexpr bool A16 = (sizeof(TA) == 2);   // f16 -> async path
  constexpr bool B16 = (sizeof(TB) == 2);
  constexpr bool ANY_ASYNC = A16 || B16;

  __shared__ __align__(16) _Float16 As[2][128][40];
  __shared__ __align__(16) _Float16 Bs[2][128][40];

  const int z = blockIdx.z;
  A += (long)z * strideAz;
  B += (long)z * strideBz;

  const int m0 = blockIdx.y * 128;
  const int n0 = blockIdx.x * 128;
  const int tid  = threadIdx.x;
  const int wave = tid >> 5;
  const int lane = tid & 31;
  const int waveM = wave >> 2;     // 0..1 -> 64 rows each
  const int waveN = wave & 3;      // 0..3 -> 32 cols each
  const int lhi = lane >> 4;       // 0/1
  const int l16 = lane & 15;

  // per-thread copy chunks: 2 x (row, 8-elem column group)
  int rr[2], cc[2];
#pragma unroll
  for (int c = 0; c < 2; ++c) {
    const int idx = c * 256 + tid;   // 0..511
    rr[c] = idx >> 2;                // 0..127
    cc[c] = (idx & 3) * 8;           // 0,8,16,24
  }

  StageChunk sA[2], sB[2];

  auto issueTile = [&](int k0, int buf) {
#pragma unroll
    for (int c = 0; c < 2; ++c) {
      const int r = rr[c], col = cc[c];
      const TA* pa = A + (long)(m0 + r) * lda + (k0 + col);
      if constexpr (A16) async_cp16B(&As[buf][r][col], pa);
      else               ld_stage(sA[c], (const float*)pa);

      const bool okB = (n0 + r) < N;
      const TB* pb = B + (long)(n0 + r) * ldb + (k0 + col);
      if constexpr (B16) {
        if (okB) async_cp16B(&Bs[buf][r][col], pb);
        // stale LDS for rows >= N only feeds epilogue-masked columns
      } else {
        if (okB) ld_stage(sB[c], (const float*)pb);
        else     zero_stage(sB[c]);
      }
    }
  };

  auto commitTile = [&](int buf) {
#pragma unroll
    for (int c = 0; c < 2; ++c) {
      const int r = rr[c], col = cc[c];
      if constexpr (!A16) *(v8h*)&As[buf][r][col] = to_h8(sA[c]);
      if constexpr (!B16) *(v8h*)&Bs[buf][r][col] = to_h8(sB[c]);
    }
  };

  v8f acc[4][2] = {};

  // ---- pipeline prologue: tile 0 -> buffer 0 ----
  issueTile(0, 0);
  commitTile(0);
  if constexpr (ANY_ASYNC) wait_asynccnt0();
  __syncthreads();

  const int ksteps = K >> 5;
  int buf = 0;
  for (int t = 0; t < ksteps; ++t) {
    const bool nxt = (t + 1) < ksteps;
    if (nxt) issueTile((t + 1) * 32, buf ^ 1);

    // ---- fragments from LDS ----
    // A (16x32 f16): lane<16 holds K {0..7,16..23}; lane>=16 holds K {8..15,24..31}
    V16 af[4];
    const int koffA = lhi * 8;
#pragma unroll
    for (int i = 0; i < 4; ++i) {
      const int r = waveM * 64 + i * 16 + l16;
      af[i].h[0] = *(const v8h*)&As[buf][r][koffA];
      af[i].h[1] = *(const v8h*)&As[buf][r][koffA + 16];
    }
    // B (32x16 f16): lane<16 holds K {0..7,16..23} of col l16; lane>=16 holds K {8..15,24..31}
    V16 bf[2];
    const int kgB = lhi * 8;
#pragma unroll
    for (int j = 0; j < 2; ++j) {
      const int r = waveN * 32 + j * 16 + l16;
      bf[j].h[0] = *(const v8h*)&Bs[buf][r][kgB];
      bf[j].h[1] = *(const v8h*)&Bs[buf][r][kgB + 16];
    }

#pragma unroll
    for (int i = 0; i < 4; ++i)
#pragma unroll
      for (int j = 0; j < 2; ++j)
        acc[i][j] = __builtin_amdgcn_wmma_f32_16x16x32_f16(
            false, af[i].v, false, bf[j].v, (short)0, acc[i][j], false, false);

    if (nxt) {
      commitTile(buf ^ 1);
      if constexpr (ANY_ASYNC) wait_asynccnt0();
    }
    __syncthreads();
    buf ^= 1;
  }

  // ---- epilogue ----
  long outOff;
  if (EPI == EPI_F16_CTX)
    outOff = (long)(z >> 4) * 2097152 + (long)(z & 15) * 64;  // b*S*1024 + h*64
  else
    outOff = (long)z * strideCz;

#pragma unroll
  for (int i = 0; i < 4; ++i) {
#pragma unroll
    for (int j = 0; j < 2; ++j) {
      const int n = n0 + waveN * 32 + j * 16 + l16;
      if (n >= N) continue;
#pragma unroll
      for (int r = 0; r < 8; ++r) {
        const int m = m0 + waveM * 64 + i * 16 + 8 * lhi + r;
        const float d = acc[i][j][r] * scale;
        if (EPI == EPI_F16_PLAIN) {
          ((half_t*)Out)[(long)m * ldo + n] = (half_t)d;
        } else if (EPI == EPI_F16_VHEAD) {
          // Pv[m,n] -> Vt[b, h][dd, s2] (per-head transposed)
          const int b  = m >> 11;            // /2048
          const int s  = m & 2047;
          const int h  = s >> 7;             // /128
          const int s2 = (s & 127) * 16 + (n >> 6);
          const int dd = n & 63;
          ((half_t*)Out)[(long)b * 2097152 + (long)h * 131072 + (long)dd * 2048 + s2] = (half_t)d;
        } else if (EPI == EPI_F32_SCORES) {
          ((float*)Out)[outOff + (long)m * ldo + n] = d;
        } else if (EPI == EPI_F16_CTX) {
          ((half_t*)Out)[outOff + (long)m * 1024 + n] = (half_t)d;
        } else { // EPI_F32_RES
          const long idx = (long)m * ldo + n;
          ((float*)Out)[idx] = d + Res[idx];
        }
      }
    }
  }
}

// In-place row softmax over 2048 columns; one block per row.
__global__ __launch_bounds__(256) void softmax_inplace(float* __restrict__ attn)
{
  __shared__ float red[256];
  const long base = (long)blockIdx.x * 2048;
  const int tid = threadIdx.x;
  float* p = attn + base + tid * 8;

  float v[8];
  *(float4*)&v[0] = *(const float4*)p;
  *(float4*)&v[4] = *(const float4*)(p + 4);

  float mx = v[0];
#pragma unroll
  for (int r = 1; r < 8; ++r) mx = fmaxf(mx, v[r]);
  red[tid] = mx;
  __syncthreads();
  for (int s = 128; s > 0; s >>= 1) {
    if (tid < s) red[tid] = fmaxf(red[tid], red[tid + s]);
    __syncthreads();
  }
  mx = red[0];
  __syncthreads();

  float sum = 0.f;
#pragma unroll
  for (int r = 0; r < 8; ++r) { v[r] = __expf(v[r] - mx); sum += v[r]; }
  red[tid] = sum;
  __syncthreads();
  for (int s = 128; s > 0; s >>= 1) {
    if (tid < s) red[tid] += red[tid + s];
    __syncthreads();
  }
  const float inv = 1.0f / red[0];

#pragma unroll
  for (int r = 0; r < 8; ++r) v[r] *= inv;
  *(float4*)p       = *(float4*)&v[0];
  *(float4*)(p + 4) = *(float4*)&v[4];
}

// LayerNorm over 1024 columns; one block per row.
__global__ __launch_bounds__(256) void layernorm_kernel(
    const float* __restrict__ x, const float* __restrict__ gamma,
    const float* __restrict__ beta, float* __restrict__ out)
{
  __shared__ float red[256];
  const long base = (long)blockIdx.x * 1024;
  const int tid = threadIdx.x;

  float v[4];
  *(float4*)v = *(const float4*)(x + base + tid * 4);

  float s = v[0] + v[1] + v[2] + v[3];
  red[tid] = s;
  __syncthreads();
  for (int st = 128; st > 0; st >>= 1) {
    if (tid < st) red[tid] += red[tid + st];
    __syncthreads();
  }
  const float mu = red[0] * (1.0f / 1024.0f);
  __syncthreads();

  float sq = 0.f;
#pragma unroll
  for (int r = 0; r < 4; ++r) { const float dv = v[r] - mu; sq += dv * dv; }
  red[tid] = sq;
  __syncthreads();
  for (int st = 128; st > 0; st >>= 1) {
    if (tid < st) red[tid] += red[tid + st];
    __syncthreads();
  }
  const float inv = rsqrtf(red[0] * (1.0f / 1024.0f) + 1e-6f);

  float o[4];
#pragma unroll
  for (int r = 0; r < 4; ++r)
    o[r] = (v[r] - mu) * inv * gamma[tid * 4 + r] + beta[tid * 4 + r];
  *(float4*)(out + base + tid * 4) = *(float4*)o;
}

extern "C" void kernel_launch(void* const* d_in, const int* in_sizes, int n_in,
                              void* d_out, int out_size, void* d_ws, size_t ws_size,
                              hipStream_t stream)
{
  (void)in_sizes; (void)n_in; (void)out_size; (void)ws_size;

  const float* q     = (const float*)d_in[0];
  const float* k     = (const float*)d_in[1];
  const float* v     = (const float*)d_in[2];
  const float* w_qs  = (const float*)d_in[3];
  const float* w_ks  = (const float*)d_in[4];
  const float* w_vs  = (const float*)d_in[5];
  const float* w_fc  = (const float*)d_in[6];
  const float* gamma = (const float*)d_in[7];
  const float* beta  = (const float*)d_in[8];

  float* out  = (float*)d_out;                 // [4,2048,1024]
  float* attn = out + 8388608ll;               // [4,16,2048,2048]

  char* w = (char*)d_ws;
  half_t* Pq  = (half_t*)(w);                  // [8192,1024] f16
  half_t* Pk  = (half_t*)(w + (1ll << 24));    // [8192,1024] f16
  half_t* Pvt = (half_t*)(w + (2ll << 24));    // [B,H][64,2048] f16 (transposed heads)
  half_t* ctx = (half_t*)(w + (3ll << 24));    // [4,2048,1024] f16
  float*  x   = (float*) (w + (4ll << 24));    // [8192,1024] f32

  const dim3 blk(256);

  // 1) Projections: P = act @ W^T   (M=8192, N=1024, K=1024)
  gemm_wmma<float, float, EPI_F16_PLAIN><<<dim3(8, 64, 1), blk, 0, stream>>>(
      q, 1024, 0, w_qs, 1024, 0, Pq, 1024, 0, nullptr, 8192, 1024, 1024, 1.0f);
  gemm_wmma<float, float, EPI_F16_PLAIN><<<dim3(8, 64, 1), blk, 0, stream>>>(
      k, 1024, 0, w_ks, 1024, 0, Pk, 1024, 0, nullptr, 8192, 1024, 1024, 1.0f);
  gemm_wmma<float, float, EPI_F16_VHEAD><<<dim3(8, 64, 1), blk, 0, stream>>>(
      v, 1024, 0, w_vs, 1024, 0, Pvt, 0, 0, nullptr, 8192, 1024, 1024, 1.0f);

  // 2) Scores per head: Q[2048,64] @ K^T -> attn (raw, scaled 1/8), z = b*16+h
  gemm_wmma<_Float16, _Float16, EPI_F32_SCORES><<<dim3(16, 16, 64), blk, 0, stream>>>(
      Pq, 64, 131072, Pk, 64, 131072, attn, 2048, 4194304, nullptr,
      2048, 2048, 64, 0.125f);

  // 3) Softmax in place inside d_out (131072 rows of 2048)
  softmax_inplace<<<131072, blk, 0, stream>>>(attn);

  // 4) ctx = attn @ V per head (A fp32 converted in regs; B = Vt [64,2048])
  gemm_wmma<float, _Float16, EPI_F16_CTX><<<dim3(1, 16, 64), blk, 0, stream>>>(
      attn, 2048, 4194304, Pvt, 2048, 131072, ctx, 1024, 0, nullptr,
      2048, 64, 2048, 1.0f);

  // 5) output = ctx @ w_fc^T + residual(q) -> x
  gemm_wmma<_Float16, float, EPI_F32_RES><<<dim3(8, 64, 1), blk, 0, stream>>>(
      ctx, 1024, 0, w_fc, 1024, 0, x, 1024, 0, q, 8192, 1024, 1024, 1.0f);

  // 6) LayerNorm -> d_out
  layernorm_kernel<<<8192, blk, 0, stream>>>(x, gamma, beta, out);
}
